// CrossAttention_44770739093803
// MI455X (gfx1250) — matrix operand order repaired and checked by reading, not compiled, not run on hardware
//
#include <hip/hip_runtime.h>

// Problem constants (match reference)
#define Bc  2
#define Cc  512
#define Nc  2048
#define Mc  2048
#define Hc  8
#define HDc 64
// SCALE (HD^-0.5) and log2(e) folded into the Q projection so attention
// scores come out of the WMMA already in log2-softmax domain.
#define QSCALEc (0.125f * 1.44269504088896340736f)

typedef __attribute__((ext_vector_type(16))) _Float16 v16h;
typedef __attribute__((ext_vector_type(8)))  _Float16 v8h;
typedef __attribute__((ext_vector_type(2)))  _Float16 v2h;
typedef __attribute__((ext_vector_type(8)))  float    v8f;

// ---- WMMA (wave32, 16x16x32 f16 -> f32 accumulate) -------------------------
__device__ __forceinline__ v8f wmma_f16(v16h a, v16h b, v8f c) {
    return __builtin_amdgcn_wmma_f32_16x16x32_f16(
        /*neg_a=*/false, a, /*neg_b=*/false, b,
        /*c_mod=*/(short)0, c, /*reuse_a=*/false, /*reuse_b=*/false);
}

// Build a v16h fragment from two 16-byte (8 halfword) vector loads.
__device__ __forceinline__ v16h frag2(const _Float16* p0, const _Float16* p1) {
    v8h lo = *(const v8h*)p0;
    v8h hi = *(const v8h*)p1;
    return __builtin_shufflevector(lo, hi, 0, 1, 2, 3, 4, 5, 6, 7, 8, 9, 10,
                                   11, 12, 13, 14, 15);
}

__device__ __forceinline__ v2h pack2(float a, float b) {
    v2h p;
    p[0] = (_Float16)a;
    p[1] = (_Float16)b;
    return p;
}

// ---- DPP 16-lane row reduce-broadcast: 1 VALU instruction per step ---------
// src0 carries the DPP read; result = op(dpp(x), x).
#define DEF_DPP_STEP(name, inst, ctrl)                                         \
    __device__ __forceinline__ float name(float x) {                           \
        float d;                                                               \
        asm(inst " %0, %1, %1 " ctrl                                           \
                 " row_mask:0xf bank_mask:0xf bound_ctrl:1"                    \
            : "=v"(d)                                                          \
            : "v"(x));                                                         \
        return d;                                                              \
    }
DEF_DPP_STEP(max_x1, "v_max_num_f32", "quad_perm:[1,0,3,2]")
DEF_DPP_STEP(max_x2, "v_max_num_f32", "quad_perm:[2,3,0,1]")
DEF_DPP_STEP(max_x4, "v_max_num_f32", "row_half_mirror")
DEF_DPP_STEP(max_x8, "v_max_num_f32", "row_mirror")
DEF_DPP_STEP(add_x1, "v_add_f32", "quad_perm:[1,0,3,2]")
DEF_DPP_STEP(add_x2, "v_add_f32", "quad_perm:[2,3,0,1]")
DEF_DPP_STEP(add_x4, "v_add_f32", "row_half_mirror")
DEF_DPP_STEP(add_x8, "v_add_f32", "row_mirror")

__device__ __forceinline__ float row16_max(float x) {
    return max_x8(max_x4(max_x2(max_x1(x))));
}
__device__ __forceinline__ float row16_sum(float x) {
    return add_x8(add_x4(add_x2(add_x1(x))));
}

// ---- Generic tiled GEMM: out[row, col] = A[row, :] @ W[:, col] -------------
// Block: 256 threads (8 waves), 256-row x 64-col tile; wave = 32 rows.
// MODE 0: A = x^T (f32, strided [C,N]), out = Q f16 [B,N,C] scaled by QSCALE
// MODE 1: A = y^T (f32, strided [C,M]), W = Wkv [C,2C]; even cols -> K f16,
//         odd cols -> V f16 with fused v*(1+lw)+lb
// MODE 2: A = core (f16 [B,N,C], async-staged), out = f32 d_out [B,C,N] + bias
template <int MODE>
__global__ __launch_bounds__(256) void gemm_wmma_kernel(
    const void* __restrict__ Aall, const float* __restrict__ W,
    const float* __restrict__ lw, const float* __restrict__ lb,
    const float* __restrict__ bias, void* __restrict__ out0,
    void* __restrict__ out1) {
    constexpr int NN  = (MODE == 1) ? (2 * Cc) : Cc;
    constexpr int ALD = 40;   // 80 B rows: 16B-aligned
    constexpr int BFP = 24;   // fragment row stride (48 B)
    __shared__ __attribute__((aligned(16))) _Float16 As[256 * ALD];
    __shared__ __attribute__((aligned(16))) _Float16 Bf[128 * BFP];

    const int tid  = threadIdx.x;
    const int lane = tid & 31;
    const int wave = tid >> 5;
    const int b    = blockIdx.z;
    const int row0 = blockIdx.y * 256;
    const int col0 = blockIdx.x * 64;
    const int sub  = lane >> 4, r = lane & 15;

    v8f acc[2][4];
#pragma unroll
    for (int t = 0; t < 2; ++t)
#pragma unroll
        for (int i = 0; i < 4; ++i)
#pragma unroll
            for (int e = 0; e < 8; ++e) acc[t][i][e] = 0.0f;

    for (int k0 = 0; k0 < Cc; k0 += 32) {
        // ---- stage A tile (256 rows x 32 K) ----
        if constexpr (MODE == 2) {
            // contiguous f16 source: async global->LDS, 16B chunks
            const _Float16* A = (const _Float16*)Aall + (size_t)b * Nc * Cc;
#pragma unroll
            for (int e = 0; e < 4; ++e) {
                int idx = tid + e * 256;   // 0..1023
                int rr  = idx >> 2;
                int ch  = idx & 3;
                unsigned lds = (unsigned)(uintptr_t)(As + rr * ALD + ch * 8);
                const _Float16* src =
                    A + (size_t)(row0 + rr) * Cc + k0 + ch * 8;
                asm volatile("global_load_async_to_lds_b128 %0, %1, off" ::"v"(
                                 lds),
                             "v"((unsigned long long)(uintptr_t)src)
                             : "memory");
            }
        } else {
            // transpose-convert: pack k-pairs -> one ds_store_b32
            const float* A = (const float*)Aall + (size_t)b * Cc * Nc;
#pragma unroll
            for (int e = 0; e < 16; ++e) {
                int idx = tid + e * 256;   // 0..4095 pairs
                int rr  = idx & 255;
                int kp  = idx >> 8;        // 0..15
                float a0 = A[(size_t)(k0 + 2 * kp) * Nc + row0 + rr];
                float a1 = A[(size_t)(k0 + 2 * kp + 1) * Nc + row0 + rr];
                *(v2h*)(As + rr * ALD + 2 * kp) = pack2(a0, a1);
            }
        }
        // ---- stage B tile (32 K x 64 cols), fragment-contiguous layout ----
#pragma unroll
        for (int e = 0; e < 4; ++e) {
            int idx  = tid + e * 256;      // 0..1023 pairs
            int kp   = idx >> 6;           // 0..15
            int cc   = idx & 63;
            int subk = kp >> 3;            // (2kp) >> 4
            int j0   = (2 * kp) & 15;
            int cs   = cc >> 4, rc = cc & 15;
            float b0 = W[(size_t)(k0 + 2 * kp) * NN + col0 + cc];
            float b1 = W[(size_t)(k0 + 2 * kp + 1) * NN + col0 + cc];
            *(v2h*)(Bf + ((cs * 2 + subk) * 16 + rc) * BFP + j0) =
                pack2(b0, b1);
        }
        if constexpr (MODE == 2)
            asm volatile("s_wait_asynccnt 0x0" ::: "memory");
        __syncthreads();

        v16h a[2];
#pragma unroll
        for (int rt = 0; rt < 2; ++rt) {
            const _Float16* ap =
                As + (wave * 32 + rt * 16 + r) * ALD + 16 * sub;
            a[rt] = frag2(ap, ap + 8);
        }
#pragma unroll
        for (int cs = 0; cs < 4; ++cs) {
            const _Float16* bp2 = Bf + ((cs * 2 + sub) * 16 + r) * BFP;
            v16h bf = frag2(bp2, bp2 + 8);   // reused across both row tiles
#pragma unroll
            for (int rt = 0; rt < 2; ++rt)
                acc[rt][cs] = wmma_f16(a[rt], bf, acc[rt][cs]);
        }
        __syncthreads();
    }

    // ---- epilogue ----
#pragma unroll
    for (int rt = 0; rt < 2; ++rt) {
#pragma unroll
        for (int cs = 0; cs < 4; ++cs) {
#pragma unroll
            for (int i = 0; i < 8; ++i) {
                int row = row0 + wave * 32 + rt * 16 + sub * 8 + i;
                int col = col0 + cs * 16 + r;
                float v = acc[rt][cs][i];
                if constexpr (MODE == 0) {
                    // fold softmax scale + log2(e) into Q
                    ((_Float16*)out0)[((size_t)b * Nc + row) * Cc + col] =
                        (_Float16)(v * QSCALEc);
                } else if constexpr (MODE == 1) {
                    int c = col >> 1;      // channel = h*HD + d
                    if ((col & 1) == 0) {
                        ((_Float16*)out0)[((size_t)b * Mc + row) * Cc + c] =
                            (_Float16)v;   // K
                    } else {
                        float vv = v * (1.0f + lw[c]) + lb[c];
                        ((_Float16*)out1)[((size_t)b * Mc + row) * Cc + c] =
                            (_Float16)vv;  // V (fused depthwise)
                    }
                } else {
                    ((float*)out0)[((size_t)b * Cc + col) * Nc + row] =
                        v + bias[col];
                }
            }
        }
    }
}

// ---- Flash attention: softmax(Q K^T * s) @ V, per (b,h) --------------------
// 8 waves x 16 Q-rows; 64 key columns per iteration (16 WMMAs per step).
// Scores arrive pre-scaled (log2 domain); softmax uses exp2 directly.
__global__ __launch_bounds__(256) void attn_wmma_kernel(
    const _Float16* __restrict__ Q, const _Float16* __restrict__ Kg,
    const _Float16* __restrict__ Vg, _Float16* __restrict__ core) {
    constexpr int KLD = 72;   // 144 B rows (16B-aligned)
    constexpr int PLD = 72;   // 64 cols + pad
    constexpr int VFP = 24;   // fragment-contiguous V rows (48 B)
    __shared__ __attribute__((aligned(16))) _Float16 Kt[64 * KLD];
    __shared__ __attribute__((aligned(16))) _Float16 Vf[256 * VFP];
    __shared__ __attribute__((aligned(16))) _Float16 Pl[8][16 * PLD];

    const int tid  = threadIdx.x;
    const int lane = tid & 31;
    const int wave = tid >> 5;
    const int b    = blockIdx.z;
    const int h    = blockIdx.y;
    const int n0   = blockIdx.x * 128 + wave * 16;
    const int sub  = lane >> 4, r = lane & 15;

    const _Float16* Qh = Q  + (size_t)b * Nc * Cc + (size_t)h * HDc;
    const _Float16* Kh = Kg + (size_t)b * Mc * Cc + (size_t)h * HDc;
    const _Float16* Vh = Vg + (size_t)b * Mc * Cc + (size_t)h * HDc;

    // Q A-fragments, d = 0..31 / 32..63: four global_load_b128 per lane
    const _Float16* qp = Qh + (size_t)(n0 + r) * Cc + 16 * sub;
    v16h qa0 = frag2(qp, qp + 8);
    v16h qa1 = frag2(qp + 32, qp + 40);

    v8f o[4];
#pragma unroll
    for (int i = 0; i < 4; ++i)
#pragma unroll
        for (int e = 0; e < 8; ++e) o[i][e] = 0.0f;
    float mrow[8], lrow[8];
#pragma unroll
    for (int i = 0; i < 8; ++i) { mrow[i] = -1e30f; lrow[i] = 0.0f; }

    for (int m0 = 0; m0 < Mc; m0 += 64) {
        __syncthreads();   // previous iteration's tile reads complete

        // ---- K tile 64x64: async global->LDS (ASYNCcnt), 2x16B/thread ----
#pragma unroll
        for (int e = 0; e < 2; ++e) {
            int idx = tid + e * 256;       // 0..511
            int rr  = idx >> 3;
            int ch  = idx & 7;
            unsigned lds = (unsigned)(uintptr_t)(Kt + rr * KLD + ch * 8);
            const _Float16* src = Kh + (size_t)(m0 + rr) * Cc + ch * 8;
            asm volatile("global_load_async_to_lds_b128 %0, %1, off" ::"v"(
                             lds),
                         "v"((unsigned long long)(uintptr_t)src)
                         : "memory");
        }
        // ---- V tile 64x64 into fragment-contiguous layout, b32 stores ----
        {
            int pr = tid >> 3;             // row pair 0..31 -> m {2pr, 2pr+1}
            int ch = tid & 7;              // d chunk of 8
            int m1 = 2 * pr;
            v8h va = *(const v8h*)(Vh + (size_t)(m0 + m1) * Cc + ch * 8);
            v8h vb2 = *(const v8h*)(Vh + (size_t)(m0 + m1 + 1) * Cc + ch * 8);
            int subk = m1 >> 4, j0 = m1 & 15;   // j0 even
#pragma unroll
            for (int e = 0; e < 8; ++e) {
                int d  = ch * 8 + e;
                int rc = d & 15, dc = d >> 4;
                v2h p;
                p[0] = va[e];
                p[1] = vb2[e];
                *(v2h*)(Vf + ((dc * 4 + subk) * 16 + rc) * VFP + j0) = p;
            }
        }
        asm volatile("s_wait_asynccnt 0x0" ::: "memory");
        __syncthreads();

        // ---- scores S[16 x 64] = Q @ K^T : four 16x16 C-tiles -------------
        v8f sfr[4];
#pragma unroll
        for (int ms = 0; ms < 4; ++ms) {
            v8f s;
#pragma unroll
            for (int e = 0; e < 8; ++e) s[e] = 0.0f;
#pragma unroll
            for (int d0 = 0; d0 < 64; d0 += 32) {
                const _Float16* kp = Kt + (ms * 16 + r) * KLD + d0 + 16 * sub;
                v16h kb = frag2(kp, kp + 8);
                s = wmma_f16(d0 ? qa1 : qa0, kb, s);
            }
            sfr[ms] = s;
        }

        // ---- online softmax: log2 domain, exp2 == v_exp_f32 ---------------
        _Float16* pw = &Pl[wave][0];
#pragma unroll
        for (int i = 0; i < 8; ++i) {
            float s0 = sfr[0][i];
            float s1 = sfr[1][i];
            float s2 = sfr[2][i];
            float s3 = sfr[3][i];
            float mx = row16_max(fmaxf(fmaxf(s0, s1), fmaxf(s2, s3)));
            float mnew = fmaxf(mrow[i], mx);
            float f  = exp2f(mrow[i] - mnew);
            float p0 = exp2f(s0 - mnew);
            float p1 = exp2f(s1 - mnew);
            float p2 = exp2f(s2 - mnew);
            float p3 = exp2f(s3 - mnew);
            float ps = row16_sum((p0 + p1) + (p2 + p3));
            lrow[i] = lrow[i] * f + ps;
            mrow[i] = mnew;
            int prow = (sub * 8 + i) * PLD + r;
            pw[prow]      = (_Float16)p0;
            pw[prow + 16] = (_Float16)p1;
            pw[prow + 32] = (_Float16)p2;
            pw[prow + 48] = (_Float16)p3;
            o[0][i] *= f; o[1][i] *= f; o[2][i] *= f; o[3][i] *= f;
        }

        // ---- P (16x64) @ V (64x64): 8 WMMAs into running accumulators -----
        const _Float16* pp = pw + r * PLD + 16 * sub;
        v16h pa0 = frag2(pp, pp + 8);
        v16h pa1 = frag2(pp + 32, pp + 40);
#pragma unroll
        for (int dc = 0; dc < 4; ++dc) {
#pragma unroll
            for (int kh = 0; kh < 2; ++kh) {
                const _Float16* vp =
                    Vf + ((dc * 4 + kh * 2 + sub) * 16 + r) * VFP;
                v16h vb = frag2(vp, vp + 8);
                o[dc] = wmma_f16(kh ? pa1 : pa0, vb, o[dc]);
            }
        }
    }

    // ---- finalize: divide by row sums, store f16 core [B, N, C] -----------
#pragma unroll
    for (int i = 0; i < 8; ++i) {
        float inv = 1.0f / lrow[i];
        int row   = n0 + sub * 8 + i;
#pragma unroll
        for (int dc = 0; dc < 4; ++dc) {
            int col = h * HDc + dc * 16 + r;
            core[((size_t)b * Nc + row) * Cc + col] =
                (_Float16)(o[dc][i] * inv);
        }
    }
}

extern "C" void kernel_launch(void* const* d_in, const int* in_sizes, int n_in,
                              void* d_out, int out_size, void* d_ws,
                              size_t ws_size, hipStream_t stream) {
    const float* x   = (const float*)d_in[0];
    const float* y   = (const float*)d_in[1];
    const float* Wq  = (const float*)d_in[2];
    const float* Wkv = (const float*)d_in[3];
    const float* lw  = (const float*)d_in[4];
    const float* lb  = (const float*)d_in[5];
    const float* Wp  = (const float*)d_in[6];
    const float* bp  = (const float*)d_in[7];
    float* out = (float*)d_out;

    const size_t tsz = (size_t)Bc * Nc * Cc;
    _Float16* Qf   = (_Float16*)d_ws;
    _Float16* Kf   = Qf + tsz;
    _Float16* Vf   = Kf + tsz;
    _Float16* core = Vf + tsz;

    dim3 blk(256);
    gemm_wmma_kernel<0><<<dim3(Cc / 64, Nc / 256, Bc), blk, 0, stream>>>(
        x, Wq, nullptr, nullptr, nullptr, Qf, nullptr);
    gemm_wmma_kernel<1><<<dim3(2 * Cc / 64, Mc / 256, Bc), blk, 0, stream>>>(
        y, Wkv, lw, lb, nullptr, Kf, Vf);
    attn_wmma_kernel<<<dim3(Nc / 128, Hc, Bc), blk, 0, stream>>>(Qf, Kf, Vf,
                                                                 core);
    gemm_wmma_kernel<2><<<dim3(Cc / 64, Nc / 256, Bc), blk, 0, stream>>>(
        core, Wp, nullptr, nullptr, bp, out, nullptr);
}
